// GroupedGRULayer_29557964931364
// MI455X (gfx1250) — matrix operand order repaired and checked by reading, not compiled, not run on hardware
//
#include <hip/hip_runtime.h>
#include <cstddef>

// Grouped GRU: G=8 groups, H=I=64, B=32, T derived at launch.
// One persistent workgroup per group; h kept in LDS; WMMA f16 -> f32.
// 16 waves; each wave owns one r/z tile (K=128) + one n-gate tile (K=64):
// 6 WMMAs per wave per step, balanced, uniform code path.

#define NG 8
#define NH 64
#define NI 64
#define NB 32
#define AST 72     // LDS A-buffer row stride (halves): 144B rows -> conflict-free b128 reads
#define SST 260    // LDS gate-staging row stride (floats)

typedef __attribute__((ext_vector_type(16))) _Float16 v16h;
typedef __attribute__((ext_vector_type(8)))  _Float16 v8h;
typedef __attribute__((ext_vector_type(8)))  float    v8f;

__device__ __forceinline__ v8f wmma16(v16h a, v16h b, v8f c) {
  // (neg_a, A, neg_b, B, c_mod, C, reuse_a, reuse_b)
  return __builtin_amdgcn_wmma_f32_16x16x32_f16(false, a, false, b, (short)0, c, false, false);
}

// 16 consecutive f32 from global -> v16h (B fragment row segment, K ascending)
__device__ __forceinline__ v16h load16f(const float* __restrict__ p) {
  v16h r;
#pragma unroll
  for (int e = 0; e < 16; ++e) r[e] = (_Float16)p[e];
  return r;
}

// A fragment, contiguous K order: lane<16 -> K = kc..kc+15
// (kc already includes the +16 for lanes 16..31). Two 16B LDS reads.
__device__ __forceinline__ v16h loadA(const _Float16* row, int kc) {
  v8h lo = *reinterpret_cast<const v8h*>(row + kc);
  v8h hi = *reinterpret_cast<const v8h*>(row + kc + 8);
  v16h r;
#pragma unroll
  for (int e = 0; e < 8; ++e) { r[e] = lo[e]; r[8 + e] = hi[e]; }
  return r;
}

__device__ __forceinline__ float sigf(float v)   { return 1.0f / (1.0f + __expf(-v)); }
__device__ __forceinline__ float tanhf_(float v) { return 1.0f - 2.0f / (1.0f + __expf(2.0f * v)); }

__global__ void __launch_bounds__(512, 1)
grouped_gru_29557964931364(const float* __restrict__ x, const float* __restrict__ h0,
                           const float* __restrict__ Wih, const float* __restrict__ Whh,
                           const float* __restrict__ bih, const float* __restrict__ bhh,
                           float* __restrict__ out, int T)
{
  __shared__ __attribute__((aligned(16))) _Float16 sXA[2][NB][AST]; // x_t as f16, double-buffered
  __shared__ __attribute__((aligned(16))) _Float16 sHA[NB][AST];    // h as f16 (A operand)
  __shared__ __attribute__((aligned(16))) float    sHf[NB][NH];     // h master copy, f32
  __shared__ __attribute__((aligned(16))) float    sS[NB][SST];     // gate pre-activations
  __shared__ float sBih[192];
  __shared__ float sBhh[192];

  const int g    = blockIdx.x;
  const int tid  = threadIdx.x;
  const int wv   = tid >> 5;
  const int lane = tid & 31;
  const int lr   = lane & 15;
  const bool hiH = lane >= 16;

  const float* Wih_g = Wih + (size_t)g * 192 * 64;
  const float* Whh_g = Whh + (size_t)g * 192 * 64;

  // Wave -> tile mapping (balanced): wave w = (m, i), m = w>>3, i = w&7.
  //   r/z tile : M-tile m, N cols i*16..i*16+15 of gates j in [0,128): K=128 (x+h)
  //   n tile   : i<4 -> xn tile (x-side of n gate), cols 128+ni*16 of sS, weights Wih rows 128+ni*16
  //              i>=4 -> gn tile (h-side of n gate), cols 192+ni*16 of sS, weights Whh rows 128+ni*16
  const int  m    = wv >> 3;
  const int  i    = wv & 7;
  const bool useX = (i < 4);
  const int  ni   = i & 3;
  const int  jrz  = i * 16;          // weight-row base for r/z tile
  const int  jn   = 128 + ni * 16;   // weight-row base for n tile
  const int  scbn = (useX ? 128 : 192) + ni * 16;

  // ---- load B fragments (weights) global -> registers, f32 -> f16 ----
  // B layout: lane<16: N=lr, K=0..15 ; lane>=16: N=lr, K=16..31 (per K-slice of 32)
  const int kh = hiH ? 16 : 0;
  v16h Brz[4], Bn[2];
#pragma unroll
  for (int s = 0; s < 2; ++s) {
    Brz[s]     = load16f(Wih_g + (size_t)(jrz + lr) * 64 + s * 32 + kh);   // x slices
    Brz[2 + s] = load16f(Whh_g + (size_t)(jrz + lr) * 64 + s * 32 + kh);   // h slices
  }
  {
    const float* Wn = useX ? Wih_g : Whh_g;
#pragma unroll
    for (int s = 0; s < 2; ++s)
      Bn[s] = load16f(Wn + (size_t)(jn + lr) * 64 + s * 32 + kh);
  }

  // ---- prologue: biases, h0, x_0 ----
  if (tid < 192) { sBih[tid] = bih[g * 192 + tid]; sBhh[tid] = bhh[g * 192 + tid]; }
#pragma unroll
  for (int q = 0; q < 4; ++q) {
    int e = tid + q * 512;            // 0..2047 over [b][hc]
    int b = e >> 6, hc = e & 63;
    float hv = h0[((size_t)g * NB + b) * NH + hc];
    sHf[b][hc] = hv;
    sHA[b][hc] = (_Float16)hv;
  }
  const int bb = tid >> 4;            // x staging: row 0..31
  const int c4 = (tid & 15) << 2;     // col 0,4,...,60
  const float* xrow = x + (size_t)bb * T * (NG * NI) + (size_t)g * NI + c4;
  {
    float4 r0 = *(const float4*)(xrow);   // t = 0
    _Float16* d = &sXA[0][bb][c4];
    d[0] = (_Float16)r0.x; d[1] = (_Float16)r0.y; d[2] = (_Float16)r0.z; d[3] = (_Float16)r0.w;
  }
  __syncthreads();

  const int arow = m * 16 + lr;       // A fragment LDS row for this lane
  const int kc0 = hiH ? 16 : 0;
  int buf = 0;

  for (int t = 0; t < T; ++t) {
    // stream next x tile (latency hidden behind WMMAs)
    float4 rx = make_float4(0.f, 0.f, 0.f, 0.f);
    const bool pf = (t + 1 < T);
    if (pf) rx = *(const float4*)(xrow + (size_t)(t + 1) * (NG * NI));
    if (t + 2 < T) __builtin_prefetch(xrow + (size_t)(t + 2) * (NG * NI), 0, 1);

    // ---- GEMM: issue all fragment loads, then chain WMMAs ----
    const _Float16* xbase = &sXA[buf][arow][0];
    const _Float16* hbase = &sHA[arow][0];
    const _Float16* nbase = useX ? xbase : hbase;  // address select, not register select

    v16h Ax0 = loadA(xbase, kc0);
    v16h Ax1 = loadA(xbase, 32 + kc0);
    v16h Ah0 = loadA(hbase, kc0);
    v16h Ah1 = loadA(hbase, 32 + kc0);
    v16h An0 = loadA(nbase, kc0);
    v16h An1 = loadA(nbase, 32 + kc0);

    v8f Crz = {};
    Crz = wmma16(Ax0, Brz[0], Crz);
    Crz = wmma16(Ax1, Brz[1], Crz);
    Crz = wmma16(Ah0, Brz[2], Crz);
    Crz = wmma16(Ah1, Brz[3], Crz);

    v8f Cn = {};
    Cn = wmma16(An0, Bn[0], Cn);
    Cn = wmma16(An1, Bn[1], Cn);

    // ---- stage gate pre-activations to LDS (C/D layout: VGPR r -> M=r(+8), N=lane%16) ----
#pragma unroll
    for (int r = 0; r < 8; ++r) {
      int row = m * 16 + r + (hiH ? 8 : 0);
      sS[row][jrz  + lr] = Crz[r];
      sS[row][scbn + lr] = Cn[r];
    }

    // commit next x tile into alternate A buffer
    if (pf) {
      _Float16* d = &sXA[buf ^ 1][bb][c4];
      d[0] = (_Float16)rx.x; d[1] = (_Float16)rx.y; d[2] = (_Float16)rx.z; d[3] = (_Float16)rx.w;
    }
    __syncthreads();

    // ---- elementwise gates + state update + output ----
#pragma unroll
    for (int q = 0; q < 4; ++q) {
      int e = tid + q * 512;
      int b = e >> 6, hc = e & 63;
      float rg = sigf(sS[b][hc]      + sBih[hc]      + sBhh[hc]);
      float zg = sigf(sS[b][64 + hc] + sBih[64 + hc] + sBhh[64 + hc]);
      float xn = sS[b][128 + hc] + sBih[128 + hc];
      float gn = sS[b][192 + hc] + sBhh[128 + hc];
      float ng = tanhf_(xn + rg * gn);
      float ho = sHf[b][hc];
      float hn = (1.0f - zg) * ng + zg * ho;
      sHf[b][hc] = hn;
      sHA[b][hc] = (_Float16)hn;
      out[((size_t)b * T + t) * (NG * NH) + (size_t)g * NH + hc] = hn;
    }
    __syncthreads();
    buf ^= 1;
  }

  // ---- final hidden state: d_out tail = h_final [G,B,H] ----
  const size_t tail = (size_t)NB * T * (NG * NH);
#pragma unroll
  for (int q = 0; q < 4; ++q) {
    int e = tid + q * 512;
    int b = e >> 6, hc = e & 63;
    out[tail + ((size_t)g * NB + b) * NH + hc] = sHf[b][hc];
  }
}

extern "C" void kernel_launch(void* const* d_in, const int* in_sizes, int n_in,
                              void* d_out, int out_size, void* d_ws, size_t ws_size,
                              hipStream_t stream) {
  const float* x   = (const float*)d_in[0];
  const float* h0  = (const float*)d_in[1];
  const float* Wih = (const float*)d_in[2];
  const float* Whh = (const float*)d_in[3];
  const float* bih = (const float*)d_in[4];
  const float* bhh = (const float*)d_in[5];
  float* out = (float*)d_out;
  const int T = in_sizes[0] / (NB * NG * NI);  // x is [B,T,G*I]
  grouped_gru_29557964931364<<<dim3(NG), dim3(512), 0, stream>>>(x, h0, Wih, Whh, bih, bhh, out, T);
}
